// CustomGCN_46033459478729
// MI455X (gfx1250) — matrix operand order repaired and checked by reading, not compiled, not run on hardware
//
#include <hip/hip_runtime.h>
#include <hip/hip_bf16.h>
#include <stdint.h>

typedef __attribute__((ext_vector_type(16))) _Float16 v16h;
typedef __attribute__((ext_vector_type(8)))  float    v8f;
typedef __attribute__((ext_vector_type(8)))  _Float16 v8h;

#define NN 100000
#define EE 1600000
#define FF 128
#define HH 128
#define LL 3
#define GG 64

// ---------------------------------------------------------------------------
// Weight prep: 5 matrices (W1, W2, convW[0..2]) 128x128 f32 [k][n] row-major
// -> f16 transposed WT[mat][n][k] so each B fragment is two contiguous 16B loads.
// ---------------------------------------------------------------------------
__global__ void k_prep_w(const float* __restrict__ W1, const float* __restrict__ W2,
                         const float* __restrict__ convW, _Float16* __restrict__ wt) {
    int i = blockIdx.x * blockDim.x + threadIdx.x;
    if (i >= 5 * 128 * 128) return;
    int mat = i >> 14;
    int idx = i & 16383;
    int n = idx >> 7, k = idx & 127;
    const float* src = (mat == 0) ? W1 : (mat == 1) ? W2 : convW + (size_t)(mat - 2) * 16384;
    wt[(size_t)mat * 16384 + n * 128 + k] = (_Float16)src[k * 128 + n];
}

// ---------------------------------------------------------------------------
// WMMA GEMM: Out[N x 128] = A[N x 128] @ W[128 x 128] (+bias, +relu).
// Block = 8 waves = 64-row x 128-col strip.
//   Stage 1: async copy f32 A tile (64x128) to LDS, padded stride 132
//            (GLOBAL_LOAD_ASYNC_TO_LDS_B128, ASYNCcnt).
//   Stage 2: one cooperative repack pass converts to f16 and lays out each
//            lane's 16-half WMMA A-fragment as a contiguous 32B LDS chunk.
//   Stage 3: each wave: 4 B fragments preloaded (whole K), kt-outer/mt-inner
//            with 4 accumulators -> 16 WMMAs, 2 ds_load_b128 per WMMA, no
//            per-wave conversion, no accumulator RAW chains.
// ---------------------------------------------------------------------------
__global__ __launch_bounds__(256) void k_gemm(const float* __restrict__ A,
                                              const _Float16* __restrict__ WT,
                                              const float* __restrict__ bias,
                                              float* __restrict__ Out,
                                              int do_relu) {
    __shared__ __align__(16) float    atile[64 * 132];   // padded f32 tile, ~33 KB
    __shared__ __align__(16) _Float16 afrag[512 * 16];   // fragment-ordered f16, 16 KB
    int tid = threadIdx.x;
    int m0 = blockIdx.x * 64;

    // ---- stage 1: async global -> LDS (coalesced b128, 8 per thread) ----
    unsigned lds_atile = (unsigned)(uintptr_t)(void*)atile;  // addr[31:0] == LDS offset
#pragma unroll
    for (int j = 0; j < 8; ++j) {
        int idx = j * 256 + tid;        // float4 index in 64x32 float4 grid
        int r = idx >> 5;
        int c4 = idx & 31;
        int gr = m0 + r;
        if (gr > NN - 1) gr = NN - 1;   // clamp tail reads
        unsigned long long gp = (unsigned long long)(uintptr_t)(A + (size_t)gr * 128 + c4 * 4);
        unsigned lo = lds_atile + (unsigned)(r * 528 + c4 * 16);  // stride 132 floats
        asm volatile("global_load_async_to_lds_b128 %0, %1, off"
                     :: "v"(lo), "v"(gp) : "memory");
    }
    asm volatile("s_wait_asynccnt 0x0" ::: "memory");
    __syncthreads();

    // ---- stage 2: repack f32 tile -> f16 A fragments (2 chunks / thread) ----
    // chunk c: lane l = c&31, kt = (c>>5)&3, mt = c>>7
    // A lane layout: lanes 0-15 row m hold K = kb+0..7 and kb+16..23;
    //                lanes 16-31 row m hold K = kb+8..15 and kb+24..31.
#pragma unroll
    for (int j = 0; j < 2; ++j) {
        int c = j * 256 + tid;
        int l = c & 31, kt = (c >> 5) & 3, mt = c >> 7;
        int m = l & 15, hi = l >> 4;
        const float* s = atile + (mt * 16 + m) * 132 + kt * 32 + hi * 8;
        float4 f0 = ((const float4*)s)[0];
        float4 f1 = ((const float4*)s)[1];
        float4 f2 = ((const float4*)(s + 16))[0];
        float4 f3 = ((const float4*)(s + 16))[1];
        v16h a;
        a[0] = (_Float16)f0.x;  a[1] = (_Float16)f0.y;  a[2] = (_Float16)f0.z;  a[3] = (_Float16)f0.w;
        a[4] = (_Float16)f1.x;  a[5] = (_Float16)f1.y;  a[6] = (_Float16)f1.z;  a[7] = (_Float16)f1.w;
        a[8] = (_Float16)f2.x;  a[9] = (_Float16)f2.y;  a[10] = (_Float16)f2.z; a[11] = (_Float16)f2.w;
        a[12] = (_Float16)f3.x; a[13] = (_Float16)f3.y; a[14] = (_Float16)f3.z; a[15] = (_Float16)f3.w;
        *(v16h*)(afrag + (size_t)c * 16) = a;
    }
    __syncthreads();

    // ---- stage 3: WMMA compute ----
    int wave = tid >> 5, lane = tid & 31;
    int n0 = wave * 16;
    int hi = lane >> 4;
    int n = lane & 15;

    // preload all 4 B fragments (K = 0..127) for this wave's n-tile
    const _Float16* brow = WT + (size_t)(n0 + n) * 128 + hi * 8;
    v16h bf[4];
#pragma unroll
    for (int kt = 0; kt < 4; ++kt) bf[kt] = __builtin_shufflevector(*(const v8h*)(brow + kt * 32), *(const v8h*)(brow + kt * 32 + 16), 0, 1, 2, 3, 4, 5, 6, 7, 8, 9, 10, 11, 12, 13, 14, 15);

    v8f acc[4] = {};
#pragma unroll
    for (int kt = 0; kt < 4; ++kt) {
#pragma unroll
        for (int mt = 0; mt < 4; ++mt) {
            v16h a = *(const v16h*)(afrag + (size_t)((mt * 4 + kt) * 32 + lane) * 16);
            acc[mt] = __builtin_amdgcn_wmma_f32_16x16x32_f16(false, a, false, bf[kt],
                                                             (short)0, acc[mt], false, false);
        }
    }

    float bv = bias ? bias[n0 + n] : 0.0f;
#pragma unroll
    for (int mt = 0; mt < 4; ++mt) {
        int rowbase = m0 + mt * 16 + hi * 8;
#pragma unroll
        for (int r = 0; r < 8; ++r) {
            int gr = rowbase + r;
            if (gr < NN) {
                float v = acc[mt][r] + bv;
                if (do_relu) v = fmaxf(v, 0.0f);
                Out[(size_t)gr * 128 + n0 + n] = v;
            }
        }
    }
}

// ---------------------------------------------------------------------------
// Graph-structure kernels (layer-invariant, computed once)
// ---------------------------------------------------------------------------
__global__ void k_deg_init(float* deg) {
    int i = blockIdx.x * blockDim.x + threadIdx.x;
    if (i < NN) deg[i] = 1.0f;  // self-loop weight
}
__global__ void k_deg_edge(const int* __restrict__ col, const float* __restrict__ ew,
                           float* __restrict__ deg) {
    int e = blockIdx.x * blockDim.x + threadIdx.x;
    if (e < EE) atomicAdd(&deg[col[e]], ew[e]);
}
__global__ void k_dinv(float* __restrict__ deg, float* __restrict__ selfs) {
    int i = blockIdx.x * blockDim.x + threadIdx.x;
    if (i < NN) {
        float r = rsqrtf(deg[i]);  // deg >= 1 always (self loops)
        deg[i] = r;                // dinv in-place
        selfs[i] = r * r;          // self-loop norm
    }
}
__global__ void k_norm(const int* __restrict__ row, const int* __restrict__ col,
                       const float* __restrict__ ew, const float* __restrict__ dinv,
                       float* __restrict__ norm) {
    int e = blockIdx.x * blockDim.x + threadIdx.x;
    if (e < EE) norm[e] = dinv[row[e]] * ew[e] * dinv[col[e]];
}
__global__ void k_count(const int* __restrict__ batch, float* __restrict__ cnt) {
    int i = blockIdx.x * blockDim.x + threadIdx.x;
    if (i < NN) atomicAdd(&cnt[batch[i]], 1.0f);
}
__global__ void k_zero(float* p, int n) {
    int i = blockIdx.x * blockDim.x + threadIdx.x;
    if (i < n) p[i] = 0.0f;
}

// agg[n] = xw[n] * dinv[n]^2   (self-loop term, full init of agg), float4 wide
__global__ void k_agg_init(const float4* __restrict__ xw, const float* __restrict__ selfs,
                           float4* __restrict__ agg) {
    int i = blockIdx.x * blockDim.x + threadIdx.x;  // NH/4 threads exactly
    float s = selfs[i >> 5];
    float4 v = xw[i];
    v.x *= s; v.y *= s; v.z *= s; v.w *= s;
    agg[i] = v;
}

// Edge scatter: 32 threads per edge, float4 gather (b128) + 4 f32 atomics.
__global__ void k_edge(const int* __restrict__ row, const int* __restrict__ col,
                       const float* __restrict__ norm, const float* __restrict__ xw,
                       float* __restrict__ agg) {
    int t = blockIdx.x * blockDim.x + threadIdx.x;
    int e = t >> 5, f4 = t & 31;
    float nv = norm[e];
    float4 v = *((const float4*)xw + (size_t)row[e] * 32 + f4);
    float* dst = agg + (size_t)col[e] * 128 + f4 * 4;
    atomicAdd(dst + 0, v.x * nv);
    atomicAdd(dst + 1, v.y * nv);
    atomicAdd(dst + 2, v.z * nv);
    atomicAdd(dst + 3, v.w * nv);
}

// Fused bias + relu (in place) + pooling. batch is sorted: accumulate per run
// of equal graph-id in registers, flush on change (~32x fewer atomics).
// Post-relu values >= 0, so int-bit atomicMax == float max with 0 init.
__global__ __launch_bounds__(128) void k_finish(float* __restrict__ h,
                                                const float* __restrict__ bias,
                                                const int* __restrict__ batch,
                                                float* __restrict__ gmx,
                                                float* __restrict__ gsum) {
    int f = threadIdx.x;            // feature
    int nbase = blockIdx.x * 32;    // 32 nodes per block, N/32 = 3125 exact
    float bv = bias[f];
    int curg = batch[nbase];
    float mx = 0.0f, sm = 0.0f;
    for (int j = 0; j < 32; ++j) {
        int nid = nbase + j;
        int g = batch[nid];
        size_t idx = (size_t)nid * 128 + f;
        float v = fmaxf(h[idx] + bv, 0.0f);
        h[idx] = v;
        if (g != curg) {
            atomicMax((int*)&gmx[(size_t)curg * 128 + f], __float_as_int(mx));
            atomicAdd(&gsum[(size_t)curg * 128 + f], sm);
            curg = g; mx = 0.0f; sm = 0.0f;
        }
        mx = fmaxf(mx, v);
        sm += v;
    }
    atomicMax((int*)&gmx[(size_t)curg * 128 + f], __float_as_int(mx));
    atomicAdd(&gsum[(size_t)curg * 128 + f], sm);
}

__global__ void k_pool(const float* __restrict__ gmx, const float* __restrict__ gsum,
                       const float* __restrict__ cnt, float* __restrict__ out, int layer) {
    int i = blockIdx.x * blockDim.x + threadIdx.x;
    if (i >= GG * HH) return;
    int g = i >> 7, f = i & 127;
    float c = fmaxf(cnt[g], 1.0f);
    size_t base = (size_t)g * (LL * 2 * HH) + layer * 2 * HH;
    out[base + f] = gmx[i];
    out[base + HH + f] = gsum[i] / c;
}

// ---------------------------------------------------------------------------
extern "C" void kernel_launch(void* const* d_in, const int* in_sizes, int n_in,
                              void* d_out, int out_size, void* d_ws, size_t ws_size,
                              hipStream_t stream) {
    (void)in_sizes; (void)n_in; (void)out_size; (void)ws_size;
    const float* x     = (const float*)d_in[0];
    const float* eattr = (const float*)d_in[1];
    const float* W1    = (const float*)d_in[2];
    const float* b1    = (const float*)d_in[3];
    const float* W2    = (const float*)d_in[4];
    const float* b2    = (const float*)d_in[5];
    const float* convW = (const float*)d_in[6];
    const float* convB = (const float*)d_in[7];
    const int*   eidx  = (const int*)d_in[8];
    const int*   batch = (const int*)d_in[9];
    const int* row = eidx;
    const int* col = eidx + EE;
    float* out = (float*)d_out;

    // Workspace carve-up (256B aligned chunks)
    char* w = (char*)d_ws;
    auto carve = [&](size_t bytes) {
        char* p = w;
        w += (bytes + 255) & ~(size_t)255;
        return (void*)p;
    };
    const size_t NH = (size_t)NN * HH;
    float* bufA  = (float*)carve(sizeof(float) * NH);
    float* bufB  = (float*)carve(sizeof(float) * NH);
    float* bufC  = (float*)carve(sizeof(float) * NH);
    float* dinv  = (float*)carve(sizeof(float) * NN);   // deg then dinv in place
    float* selfs = (float*)carve(sizeof(float) * NN);
    float* norm  = (float*)carve(sizeof(float) * EE);
    float* gmx   = (float*)carve(sizeof(float) * 2 * GG * HH);  // gmx | gsum
    float* gsum  = gmx + (size_t)GG * HH;
    float* cnt   = (float*)carve(sizeof(float) * GG);
    _Float16* wt = (_Float16*)carve(sizeof(_Float16) * 5 * 16384);

    const int gemm_grid = (NN + 63) / 64;  // 1563

    // --- One-time (per call) graph structure + weight prep ---
    k_prep_w<<<(5 * 16384 + 255) / 256, 256, 0, stream>>>(W1, W2, convW, wt);
    k_zero<<<1, 64, 0, stream>>>(cnt, GG);
    k_count<<<(NN + 255) / 256, 256, 0, stream>>>(batch, cnt);
    k_deg_init<<<(NN + 255) / 256, 256, 0, stream>>>(dinv);
    k_deg_edge<<<(EE + 255) / 256, 256, 0, stream>>>(col, eattr, dinv);
    k_dinv<<<(NN + 255) / 256, 256, 0, stream>>>(dinv, selfs);
    k_norm<<<(EE + 255) / 256, 256, 0, stream>>>(row, col, eattr, dinv, norm);

    // --- MLP: h = relu(relu(x@W1+b1)@W2+b2) ---
    k_gemm<<<gemm_grid, 256, 0, stream>>>(x, wt, b1, bufA, 1);
    k_gemm<<<gemm_grid, 256, 0, stream>>>(bufA, wt + 16384, b2, bufB, 1);

    // --- 3 GCN conv layers, rotating the three N*H buffers ---
    float* h   = bufB;
    float* xw  = bufA;
    float* agg = bufC;
    for (int i = 0; i < LL; ++i) {
        k_gemm<<<gemm_grid, 256, 0, stream>>>(h, wt + (size_t)(2 + i) * 16384, nullptr, xw, 0);
        k_agg_init<<<(int)(NH / 4 / 256), 256, 0, stream>>>((const float4*)xw, selfs, (float4*)agg);
        k_edge<<<(EE * 32) / 256, 256, 0, stream>>>(row, col, norm, xw, agg);
        k_zero<<<(2 * GG * HH + 255) / 256, 256, 0, stream>>>(gmx, 2 * GG * HH);
        k_finish<<<NN / 32, 128, 0, stream>>>(agg, convB + (size_t)i * HH, batch, gmx, gsum);
        k_pool<<<(GG * HH + 255) / 256, 256, 0, stream>>>(gmx, gsum, cnt, out, i);
        float* oldh = h;
        h = agg;      // relu'd output becomes next input
        agg = xw;     // consumed xw buffer becomes next agg
        xw = oldh;    // consumed h buffer becomes next xw target
    }
}
